// LocalityAttention_78331613544746
// MI455X (gfx1250) — matrix operand order repaired and checked by reading, not compiled, not run on hardware
//
#include <hip/hip_runtime.h>
#include <hip/hip_fp16.h>

typedef __attribute__((ext_vector_type(16))) _Float16 v16h;
typedef __attribute__((ext_vector_type(8)))  float    v8f;

#define WMMA_F32_F16(A, B, C) \
    __builtin_amdgcn_wmma_f32_16x16x32_f16(false, (A), false, (B), (short)0, (C), false, false)

// ---------------------------------------------------------------------------
// Projection GEMM: Out[m, e] = sum_d X[m, d] * W[e, d] + bias[e]
// X: [rows, D] f32 row-major, W: [E, D] f32 row-major (torch Linear weight).
// Out stored as f16, either [rows, E] (TRANS=false) or [B, E, S] (TRANS=true).
// One block = 8 waves = 16 rows x 128 cols of output. The 16xD X tile is
// staged once in LDS as f16; rows padded +8 halves (16 B) so the 16 row-lanes
// rotate across all 64 LDS banks (no ds_load_b128 conflicts).
// ---------------------------------------------------------------------------
template <int D, int E, int S, bool TRANS>
__global__ __launch_bounds__(256) void proj_f16_kernel(
    const float* __restrict__ X,
    const float* __restrict__ W,
    const float* __restrict__ bias,
    _Float16* __restrict__ Out)
{
    constexpr int DP = D + 8;                  // padded row stride (halves)
    __shared__ _Float16 xs[16 * DP];           // ~24.3 KB for D=768

    const int lane = threadIdx.x & 31;
    const int wave = threadIdx.x >> 5;
    const int r    = lane & 15;                // row (A) / col (B,D) within tile
    const int hi   = lane >> 4;                // K-group select
    (void)lane;

    const int m0 = blockIdx.y * 16;
    const int n0 = blockIdx.x * 128 + wave * 16;

    // Cooperative stage of the 16 x D input tile, f32 -> f16 (padded rows).
    {
        constexpr int total = 16 * D;          // 12288 (D divisible by 4)
        for (int idx = threadIdx.x * 4; idx < total; idx += 256 * 4) {
            const int row = idx / D;           // constexpr D -> strength-reduced
            const int col = idx - row * D;
            const float* src = X + (size_t)(m0 + row) * D + col;
            _Float16* dst = xs + row * DP + col;
            dst[0] = (_Float16)src[0];
            dst[1] = (_Float16)src[1];
            dst[2] = (_Float16)src[2];
            dst[3] = (_Float16)src[3];
        }
    }
    __syncthreads();

    const _Float16* arow = xs + r * DP;
    const float*    brow = W + (size_t)(n0 + r) * D;

    v8f acc = {};
    #pragma unroll 4
    for (int kb = 0; kb < D; kb += 32) {
        v16h a, bf;
        const _Float16* ap = arow + kb + hi * 8;
        #pragma unroll
        for (int i = 0; i < 8; ++i) {
            a[i]     = ap[i];                  // K = kb + hi*8 + i
            a[8 + i] = ap[16 + i];             // K = kb + 16 + hi*8 + i
        }
        const float* bp = brow + kb + hi * 8;
        #pragma unroll
        for (int i = 0; i < 16; ++i)
            bf[i] = (_Float16)bp[(i & 7) + ((i >> 3) << 4)];   // K = kb + hi*8 + i (i<8), kb + 16 + hi*8 + i-8
        acc = WMMA_F32_F16(a, bf, acc);
    }

    const float bval = bias[n0 + r];
    #pragma unroll
    for (int r8 = 0; r8 < 8; ++r8) {
        const int row = hi * 8 + r8;           // C/D layout: M = 8*hi + vgpr
        const int gm  = m0 + row;
        const int ge  = n0 + r;
        const float v = acc[r8] + bval;
        if constexpr (!TRANS) {
            Out[(size_t)gm * E + ge] = (_Float16)v;
        } else {
            const int b = gm / S;              // S = 1024 -> shift
            const int s = gm & (S - 1);
            Out[((size_t)b * E + ge) * S + s] = (_Float16)v;   // [B,E,S]
        }
    }
}

// ---------------------------------------------------------------------------
// Fused attention: per (batch, 16-query-row tile) workgroup of 8 waves.
//  Pass 1: scores[16, S] = (Q_tile . K^T) / T, diagonal masked, into LDS.
//  Softmax (two-level LDS reduction, interleaved columns -> conflict-free),
//  unnormalized exp -> padded f16 LDS strip.
//  Pass 2: O_tile = P . V, rescaled by 1/rowsum on store.
// ---------------------------------------------------------------------------
template <int S, int E>
__global__ __launch_bounds__(256) void attn_fused_kernel(
    const _Float16* __restrict__ Qh,   // [B, S, E] f16
    const _Float16* __restrict__ Kh,   // [B, S, E] f16
    const _Float16* __restrict__ VhT,  // [B, E, S] f16 (transposed)
    const float* __restrict__ temp_ptr,
    float* __restrict__ Out)           // [B, S, E] f32
{
    constexpr int SCP = S + 4;         // score row stride (floats), +16 B skew
    constexpr int PFP = S + 8;         // prob  row stride (halves), +16 B skew

    extern __shared__ char smem[];
    float*    sc      = (float*)smem;                                  // [16][SCP]
    _Float16* pf      = (_Float16*)(smem + (size_t)16 * SCP * 4);      // [16][PFP]
    float*    red     = (float*)(smem + (size_t)16 * SCP * 4
                                      + (size_t)16 * PFP * 2);         // [16][16]
    float*    rowstat = red + 256;                                     // [16]

    const int lane = threadIdx.x & 31;
    const int wave = threadIdx.x >> 5;
    const int r    = lane & 15;
    const int hi   = lane >> 4;

    constexpr int tilesS = S >> 4;             // 64 key tiles
    const int b  = blockIdx.x / tilesS;        // constexpr divisor -> shift
    const int qt = blockIdx.x & (tilesS - 1);
    const int m0 = qt * 16;

    const float invT = 1.0f / temp_ptr[0];

    // ---------------- Pass 1: score strip into LDS ----------------
    const _Float16* qrow = Qh + ((size_t)b * S + m0 + r) * E;
    for (int jt = wave; jt < tilesS; jt += 8) {
        const int n0 = jt * 16;
        const _Float16* krow = Kh + ((size_t)b * S + n0 + r) * E;
        // Pull next tile's K rows toward the WGP while this tile computes.
        if (jt + 8 < tilesS) {
            const _Float16* nk = Kh + ((size_t)b * S + (n0 + 128) + r) * E;
            __builtin_prefetch(nk, 0, 1);
            __builtin_prefetch(nk + (E / 2), 0, 1);
        }
        v8f acc = {};
        #pragma unroll 4
        for (int kb = 0; kb < E; kb += 32) {
            v16h a, bf;
            const _Float16* ap = qrow + kb + hi * 8;
            #pragma unroll
            for (int i = 0; i < 8; ++i) { a[i] = ap[i]; a[8 + i] = ap[16 + i]; }
            const _Float16* bp = krow + kb + hi * 8;
            #pragma unroll
            for (int i = 0; i < 16; ++i) bf[i] = bp[(i & 7) + ((i >> 3) << 4)];
            acc = WMMA_F32_F16(a, bf, acc);
        }
        #pragma unroll
        for (int r8 = 0; r8 < 8; ++r8) {
            const int row = hi * 8 + r8;
            const int col = n0 + r;
            float v = acc[r8] * invT;
            if (m0 + row == col) v = -10000.0f;   // diagonal self-exclusion
            sc[row * SCP + col] = v;
        }
    }
    __syncthreads();

    // ---------------- Softmax over S per row (interleaved columns) ----------
    {
        const int row = threadIdx.x >> 4;      // 16 rows
        const int seg = threadIdx.x & 15;      // 16 segments/row
        constexpr int cn = S >> 4;             // 64 cols per thread
        const float* srow = sc + row * SCP;
        _Float16*    prow = pf + row * PFP;

        float mx = -3.0e38f;
        #pragma unroll 8
        for (int c = 0; c < cn; ++c)           // col = seg + 16*c: conflict-free
            mx = fmaxf(mx, srow[seg + 16 * c]);
        red[row * 16 + seg] = mx;
        __syncthreads();
        if (seg == 0) {
            float m2 = red[row * 16];
            #pragma unroll
            for (int i = 1; i < 16; ++i) m2 = fmaxf(m2, red[row * 16 + i]);
            rowstat[row] = m2;
        }
        __syncthreads();
        const float rm = rowstat[row];
        float sum = 0.0f;
        #pragma unroll 8
        for (int c = 0; c < cn; ++c) {
            const int col = seg + 16 * c;
            const float e = __expf(srow[col] - rm);
            prow[col] = (_Float16)e;
            sum += e;
        }
        __syncthreads();
        red[row * 16 + seg] = sum;
        __syncthreads();
        if (seg == 0) {
            float s2 = 0.0f;
            #pragma unroll
            for (int i = 0; i < 16; ++i) s2 += red[row * 16 + i];
            rowstat[row] = 1.0f / s2;
        }
        __syncthreads();
    }

    // ---------------- Pass 2: O = P . V (wave owns E/128 tiles) ----------------
    constexpr int tilesPerWave = (E >> 4) / 8;   // 6 for E=768
    #pragma unroll
    for (int t = 0; t < tilesPerWave; ++t) {
        const int e0 = (wave * tilesPerWave + t) * 16;
        const _Float16* vcol = VhT + ((size_t)b * E + e0 + r) * S;
        const _Float16* prow = pf + r * PFP;
        v8f acc = {};
        #pragma unroll 4
        for (int kb = 0; kb < S; kb += 32) {
            v16h a, bf;
            const _Float16* ap = prow + kb + hi * 8;
            #pragma unroll
            for (int i = 0; i < 8; ++i) { a[i] = ap[i]; a[8 + i] = ap[16 + i]; }
            const _Float16* bp = vcol + kb + hi * 8;
            #pragma unroll
            for (int i = 0; i < 16; ++i) bf[i] = bp[(i & 7) + ((i >> 3) << 4)];
            acc = WMMA_F32_F16(a, bf, acc);
        }
        #pragma unroll
        for (int r8 = 0; r8 < 8; ++r8) {
            const int row = hi * 8 + r8;
            Out[((size_t)b * S + m0 + row) * E + e0 + r] = acc[r8] * rowstat[row];
        }
    }
}

// ---------------------------------------------------------------------------
extern "C" void kernel_launch(void* const* d_in, const int* in_sizes, int n_in,
                              void* d_out, int out_size, void* d_ws, size_t ws_size,
                              hipStream_t stream)
{
    (void)in_sizes; (void)n_in; (void)out_size; (void)ws_size;

    const float* q  = (const float*)d_in[0];
    const float* k  = (const float*)d_in[1];
    const float* v  = (const float*)d_in[2];
    const float* Wq = (const float*)d_in[3];
    const float* bq = (const float*)d_in[4];
    const float* Wk = (const float*)d_in[5];
    const float* bk = (const float*)d_in[6];
    const float* Wv = (const float*)d_in[7];
    const float* bv = (const float*)d_in[8];
    const float* temperature = (const float*)d_in[9];
    float* out = (float*)d_out;

    constexpr int B = 16, S = 1024, D = 768, E = 768;
    constexpr int rows = B * S;

    _Float16* Qh  = (_Float16*)d_ws;                 // [B,S,E]
    _Float16* Kh  = Qh + (size_t)rows * E;           // [B,S,E]
    _Float16* VhT = Kh + (size_t)rows * E;           // [B,E,S]

    dim3 pblk(256);
    dim3 pgrd(E / 128, rows / 16);
    proj_f16_kernel<D, E, S, false><<<pgrd, pblk, 0, stream>>>(q, Wq, bq, Qh);
    proj_f16_kernel<D, E, S, false><<<pgrd, pblk, 0, stream>>>(k, Wk, bk, Kh);
    proj_f16_kernel<D, E, S, true ><<<pgrd, pblk, 0, stream>>>(v, Wv, bv, VhT);

    constexpr size_t shmem = (size_t)16 * (S + 4) * sizeof(float)      // sc
                           + (size_t)16 * (S + 8) * sizeof(_Float16)   // pf
                           + 272 * sizeof(float);                      // red+rowstat
    attn_fused_kernel<S, E><<<dim3(B * (S / 16)), dim3(256), shmem, stream>>>(
        Qh, Kh, VhT, temperature, out);
}
